// RPN_43361989820500
// MI455X (gfx1250) — matrix operand order repaired and checked by reading, not compiled, not run on hardware
//
#include <hip/hip_runtime.h>
#include <hip/hip_bf16.h>

typedef __attribute__((ext_vector_type(16))) _Float16 v16h;
typedef __attribute__((ext_vector_type(8)))  float    v8f;

#define WAVES_PER_BLOCK 8
#define PX_PER_WAVE     64
#define PX_PER_BLOCK    (WAVES_PER_BLOCK * PX_PER_WAVE)

#define HF 37
#define WF 50
#define PF (HF * WF)               // 1850
#define FEAT_ELEMS (512 * PF)      // 947200
#define NANCH (PF * 9)             // 16650
#define NSORT 32768
#define NMS_N 6000
#define POST_NMS_N 300

// ---------------------------------------------------------------------------
// f32 -> f16 convert (image)
// ---------------------------------------------------------------------------
__global__ void f32_to_f16_kernel(const float* __restrict__ in,
                                  _Float16* __restrict__ out, int n) {
  int i = blockIdx.x * blockDim.x + threadIdx.x;
  if (i < n) out[i] = (_Float16)in[i];
}

// ---------------------------------------------------------------------------
// Weight conversion: f32 [Co][K] row-major -> f16 [Cop][Kp], zero padded
// ---------------------------------------------------------------------------
__global__ void cvt_weights_kernel(const float* __restrict__ w,
                                   _Float16* __restrict__ wp,
                                   int Co, int K, int Cop, int Kp) {
  int i = blockIdx.x * blockDim.x + threadIdx.x;
  int total = Cop * Kp;
  if (i >= total) return;
  int m = i / Kp;
  int k = i - m * Kp;
  _Float16 v = (_Float16)0.0f;
  if (m < Co && k < K) v = (_Float16)w[(size_t)m * K + k];
  wp[i] = v;
}

// ---------------------------------------------------------------------------
// Implicit-GEMM conv via WMMA f32 = f16 x f16.
//   act_in : [Ci][H][W] f16       Wp : [Cop][Kp] f16 (zero padded)
//   out16  : optional f16 [Co][P] out32 : optional f32 [Co][P]
//   is3x3 : 3x3 SAME pad=1, else 1x1.
// Each wave: 32 Cout x 64 pixels (2 A-frags x 4 accums = 8 WMMA per K-step),
// B tile staged once in wave-private LDS and reused by both Cout tiles.
// Wave-local sync only (CDNA5 LDS ops are in-order within a wave; we add an
// explicit s_wait_dscnt 0 after staging) -> no block-wide barriers.
// ---------------------------------------------------------------------------
__global__ __launch_bounds__(256)
void conv_wmma_kernel(const _Float16* __restrict__ act_in,
                      const _Float16* __restrict__ Wp,
                      const float* __restrict__ bias,
                      _Float16* __restrict__ out16,
                      float* __restrict__ out32,
                      int Ci, int Co, int H, int W,
                      int K, int Kp, int is3x3, int relu) {
  __shared__ __align__(128) _Float16 lds[WAVES_PER_BLOCK * PX_PER_WAVE * 32];

  const int P      = H * W;
  const int tid    = threadIdx.x;
  const int wave   = tid >> 5;
  const int lane   = tid & 31;
  const int co0    = blockIdx.y * 32;                 // 32 Cout per block
  const int pxbase = blockIdx.x * PX_PER_BLOCK + wave * PX_PER_WAVE;
  _Float16* bt = &lds[wave * PX_PER_WAVE * 32];

  const int m      = lane & 15;        // A row / B col / D col within tile
  const int kgrp   = lane >> 4;        // 0 or 1
  const int kbaseA = kgrp * 8;         // A-lane K base within 32-chunk
  const int kbaseB = kgrp * 16;        // unused; B reads below use kbaseA (same k order as A)

  v8f acc[2][4] = {};

  const _Float16* wrow0 = Wp + (size_t)(co0 + m) * Kp + kbaseA;
  const _Float16* wrow1 = wrow0 + (size_t)16 * Kp;

  for (int k0 = 0; k0 < Kp; k0 += 32) {
    // ---- stage B tile into LDS: bt[n][k] , lane owns column k = lane ----
    {
      int k = k0 + lane;
      bool kvalid = k < K;
      int ci, dy, dx;
      if (is3x3) {
        ci = k / 9;
        int r = k - ci * 9;
        dy = r / 3 - 1;
        dx = (r - (r / 3) * 3) - 1;
      } else {
        ci = k; dy = 0; dx = 0;
      }
      const _Float16* inCh = act_in + (size_t)ci * P;
      int p  = pxbase;
      int py = p / W;
      int px = p - py * W;
#pragma unroll 4
      for (int n = 0; n < PX_PER_WAVE; ++n) {
        _Float16 v = (_Float16)0.0f;
        if (kvalid && p < P) {
          int iy = py + dy, ix = px + dx;
          if ((unsigned)iy < (unsigned)H && (unsigned)ix < (unsigned)W)
            v = inCh[iy * W + ix];
        }
        bt[n * 32 + lane] = v;
        ++p; ++px;
        if (px == W) { px = 0; ++py; }
      }
    }
    // wave-local: ensure all DS stores of this wave's tile completed before
    // the cross-lane DS reads below (LDS is in-order within a wave; this
    // makes it explicit and blocks compiler reordering).
    asm volatile("s_wait_dscnt 0x0" ::: "memory");

    // ---- A fragments: two contiguous 16B chunks of each f16 weight row ----
    union Frag { uint4 u[2]; v16h h; };
    Frag fa0, fa1;
    const _Float16* wr0 = wrow0 + k0;
    const _Float16* wr1 = wrow1 + k0;
    fa0.u[0] = *(const uint4*)(wr0);        // K = k0+kbaseA    .. +7
    fa0.u[1] = *(const uint4*)(wr0 + 16);   // K = k0+kbaseA+16 .. +23
    fa1.u[0] = *(const uint4*)(wr1);
    fa1.u[1] = *(const uint4*)(wr1 + 16);
    __builtin_prefetch(wr0 + 32, 0, 1);     // next K chunk -> global_prefetch
    __builtin_prefetch(wr1 + 32, 0, 1);

    // ---- 8 WMMA per K step (B fragment reused across both Cout tiles) ----
#pragma unroll
    for (int q = 0; q < 4; ++q) {
      Frag fb;
      const _Float16* br = &bt[(q * 16 + m) * 32 + kbaseA];
      fb.u[0] = *(const uint4*)(br);      // K = k0+kbaseA    .. +7
      fb.u[1] = *(const uint4*)(br + 16); // K = k0+kbaseA+16 .. +23
      acc[0][q] = __builtin_amdgcn_wmma_f32_16x16x32_f16(
          false, fa0.h, false, fb.h, (short)0, acc[0][q], false, false);
      acc[1][q] = __builtin_amdgcn_wmma_f32_16x16x32_f16(
          false, fa1.h, false, fb.h, (short)0, acc[1][q], false, false);
    }
    // compiler-only fence: keep next iteration's DS stores after these reads
    asm volatile("" ::: "memory");
  }

  // ---- epilogue: D layout lane = N, rows M = kgrp*8 + v ----
#pragma unroll
  for (int q = 0; q < 4; ++q) {
    int ppix = pxbase + q * 16 + m;
    if (ppix >= P) continue;
#pragma unroll
    for (int t = 0; t < 2; ++t) {
#pragma unroll
      for (int v = 0; v < 8; ++v) {
        int co = co0 + t * 16 + kgrp * 8 + v;
        if (co < Co) {
          float x = acc[t][q][v] + bias[co];
          if (relu) x = fmaxf(x, 0.0f);
          size_t off = (size_t)co * P + ppix;
          if (out16) out16[off] = (_Float16)x;
          if (out32) out32[off] = x;
        }
      }
    }
  }
}

// ---------------------------------------------------------------------------
// 2x2 stride-2 max pool (VALID), f16
// ---------------------------------------------------------------------------
__global__ void pool2_kernel(const _Float16* __restrict__ in,
                             _Float16* __restrict__ out,
                             int C, int H, int W, int Ho, int Wo) {
  int i = blockIdx.x * blockDim.x + threadIdx.x;
  int total = C * Ho * Wo;
  if (i >= total) return;
  int c = i / (Ho * Wo);
  int r = i - c * (Ho * Wo);
  int oy = r / Wo;
  int ox = r - oy * Wo;
  size_t base = (size_t)c * H * W + (size_t)(2 * oy) * W + 2 * ox;
  float a = (float)in[base];
  float b = (float)in[base + 1];
  float cc = (float)in[base + W];
  float d = (float)in[base + W + 1];
  out[i] = (_Float16)fmaxf(fmaxf(a, b), fmaxf(cc, d));
}

// ---------------------------------------------------------------------------
// fg softmax:  fg[(y*W+x)*9+a] = sigmoid(cls[9+a] - cls[a])
// ---------------------------------------------------------------------------
__global__ void fg_score_kernel(const float* __restrict__ cls,
                                float* __restrict__ fg) {
  int i = blockIdx.x * blockDim.x + threadIdx.x;
  if (i >= NANCH) return;
  int p = i / 9;
  int a = i - p * 9;
  float bg = cls[a * PF + p];
  float ff = cls[(9 + a) * PF + p];
  fg[i] = 1.0f / (1.0f + expf(bg - ff));
}

// canonical Faster-RCNN anchors (base 16, ratios .5/1/2, scales 8/16/32)
__constant__ float c_anch[9][4] = {
  {-84.f,  -40.f,  99.f,  55.f}, {-176.f,  -88.f, 191.f, 103.f},
  {-360.f, -184.f, 375.f, 199.f}, {-56.f,  -56.f,  71.f,  71.f},
  {-120.f, -120.f, 135.f, 135.f}, {-248.f, -248.f, 263.f, 263.f},
  {-36.f,  -80.f,  51.f,  95.f}, {-80.f, -168.f,  95.f, 183.f},
  {-168.f, -344.f, 183.f, 359.f}};

// ---------------------------------------------------------------------------
// box decode + clip + min-size filter; init sort keys/idx (padded to NSORT)
// ---------------------------------------------------------------------------
__global__ void decode_kernel(const float* __restrict__ fg,
                              const float* __restrict__ bbox,
                              float* __restrict__ keys,
                              int* __restrict__ sidx,
                              float* __restrict__ pbox) {
  int i = blockIdx.x * blockDim.x + threadIdx.x;
  if (i >= NSORT) return;
  sidx[i] = i;
  if (i >= NANCH) { keys[i] = -3.0e38f; return; }
  int p = i / 9;
  int a = i - p * 9;
  int y = p / WF;
  int x = p - y * WF;
  float sx = (float)x * 16.0f, sy = (float)y * 16.0f;
  float ax1 = c_anch[a][0] + sx, ay1 = c_anch[a][1] + sy;
  float ax2 = c_anch[a][2] + sx, ay2 = c_anch[a][3] + sy;
  float w = ax2 - ax1 + 1.0f, h = ay2 - ay1 + 1.0f;
  float cx = ax1 + 0.5f * w, cy = ay1 + 0.5f * h;
  float d0 = bbox[(4 * a + 0) * PF + p];
  float d1 = bbox[(4 * a + 1) * PF + p];
  float d2 = bbox[(4 * a + 2) * PF + p];
  float d3 = bbox[(4 * a + 3) * PF + p];
  float pcx = d0 * w + cx, pcy = d1 * h + cy;
  float pw = expf(d2) * w, ph = expf(d3) * h;
  float X1 = fminf(fmaxf(pcx - 0.5f * pw, 0.0f), 799.0f);
  float Y1 = fminf(fmaxf(pcy - 0.5f * ph, 0.0f), 599.0f);
  float X2 = fminf(fmaxf(pcx + 0.5f * pw, 0.0f), 799.0f);
  float Y2 = fminf(fmaxf(pcy + 0.5f * ph, 0.0f), 599.0f);
  pbox[4 * i + 0] = X1; pbox[4 * i + 1] = Y1;
  pbox[4 * i + 2] = X2; pbox[4 * i + 3] = Y2;
  bool valid = (X2 - X1 + 1.0f >= 16.0f) && (Y2 - Y1 + 1.0f >= 16.0f);
  keys[i] = valid ? fg[i] : -3.0e38f;
}

// ---------------------------------------------------------------------------
// bitonic sort step (descending by key, payload idx)
// ---------------------------------------------------------------------------
__global__ void bitonic_kernel(float* __restrict__ keys,
                               int* __restrict__ sidx, int j, int k) {
  unsigned i = blockIdx.x * blockDim.x + threadIdx.x;
  unsigned ixj = i ^ (unsigned)j;
  if (ixj <= i || ixj >= NSORT) return;
  bool desc = ((i & (unsigned)k) == 0);
  float a = keys[i], b = keys[ixj];
  bool doSwap = desc ? (a < b) : (a > b);
  if (doSwap) {
    keys[i] = b; keys[ixj] = a;
    int t = sidx[i]; sidx[i] = sidx[ixj]; sidx[ixj] = t;
  }
}

// ---------------------------------------------------------------------------
// greedy NMS over top-6000 sorted boxes, one workgroup
// ---------------------------------------------------------------------------
__global__ __launch_bounds__(1024)
void nms_kernel(const float* __restrict__ keys,
                const int* __restrict__ sidx,
                const float* __restrict__ pbox,
                unsigned char* __restrict__ keep) {
  __shared__ unsigned char supp[NMS_N];
  __shared__ float bi[4];
  __shared__ int keep_flag;
  int tid = threadIdx.x;
  for (int i = tid; i < NMS_N; i += 1024) supp[i] = 0;
  __syncthreads();
  for (int i = 0; i < NMS_N; ++i) {
    if (tid == 0) {
      int ki = (keys[i] > -1.0e30f) && (supp[i] == 0);
      keep[i] = (unsigned char)ki;
      keep_flag = ki;
      const float* b = pbox + 4 * (size_t)sidx[i];
      bi[0] = b[0]; bi[1] = b[1]; bi[2] = b[2]; bi[3] = b[3];
    }
    __syncthreads();
    if (keep_flag) {
      float x1 = bi[0], y1 = bi[1], x2 = bi[2], y2 = bi[3];
      float ai = (x2 - x1 + 1.0f) * (y2 - y1 + 1.0f);
      for (int j = i + 1 + tid; j < NMS_N; j += 1024) {
        if (supp[j]) continue;
        const float* b = pbox + 4 * (size_t)sidx[j];
        float xx1 = fmaxf(x1, b[0]), yy1 = fmaxf(y1, b[1]);
        float xx2 = fminf(x2, b[2]), yy2 = fminf(y2, b[3]);
        float iw = fmaxf(0.0f, xx2 - xx1 + 1.0f);
        float ih = fmaxf(0.0f, yy2 - yy1 + 1.0f);
        float inter = iw * ih;
        float aj = (b[2] - b[0] + 1.0f) * (b[3] - b[1] + 1.0f);
        if (inter / (ai + aj - inter) > 0.7f) supp[j] = 1;
      }
    }
    __syncthreads();
  }
}

// ---------------------------------------------------------------------------
// compact kept boxes into 300 ROIs, write im_info
// ---------------------------------------------------------------------------
__global__ void finalize_kernel(const unsigned char* __restrict__ keep,
                                const int* __restrict__ sidx,
                                const float* __restrict__ pbox,
                                float* __restrict__ out) {
  if (threadIdx.x != 0 || blockIdx.x != 0) return;
  float* rois = out + FEAT_ELEMS;
  int cnt = 0;
  for (int i = 0; i < NMS_N && cnt < POST_NMS_N; ++i) {
    if (keep[i]) {
      const float* b = pbox + 4 * (size_t)sidx[i];
      rois[cnt * 5 + 0] = 0.0f;
      rois[cnt * 5 + 1] = b[0];
      rois[cnt * 5 + 2] = b[1];
      rois[cnt * 5 + 3] = b[2];
      rois[cnt * 5 + 4] = b[3];
      ++cnt;
    }
  }
  for (; cnt < POST_NMS_N; ++cnt)
    for (int c = 0; c < 5; ++c) rois[cnt * 5 + c] = 0.0f;
  float* info = out + FEAT_ELEMS + POST_NMS_N * 5;
  info[0] = 600.0f; info[1] = 800.0f; info[2] = 1.0f;
}

// ---------------------------------------------------------------------------
// host orchestration
// ---------------------------------------------------------------------------
static inline size_t align_up_sz(size_t x) { return (x + 255) & ~(size_t)255; }

extern "C" void kernel_launch(void* const* d_in, const int* in_sizes, int n_in,
                              void* d_out, int out_size, void* d_ws, size_t ws_size,
                              hipStream_t stream) {
  (void)in_sizes; (void)n_in; (void)out_size; (void)ws_size;

  const float* image   = (const float*)d_in[0];
  const float* vw[13];
  const float* vb[13];
  for (int i = 0; i < 13; ++i) { vw[i] = (const float*)d_in[2 + i]; vb[i] = (const float*)d_in[15 + i]; }
  const float* conv1_w = (const float*)d_in[28];
  const float* conv1_b = (const float*)d_in[29];
  const float* score_w = (const float*)d_in[30];
  const float* score_b = (const float*)d_in[31];
  const float* bbox_w  = (const float*)d_in[32];
  const float* bbox_b  = (const float*)d_in[33];
  float* out = (float*)d_out;

  const size_t ACT_ELEMS = (size_t)64 * 600 * 800;   // 30.72M (max activation)
  char* ws = (char*)d_ws;
  size_t off = 0;
  auto carve = [&](size_t bytes) -> void* {
    void* p = (void*)(ws + off);
    off += align_up_sz(bytes);
    return p;
  };
  _Float16* actA   = (_Float16*)carve(ACT_ELEMS * 2);
  _Float16* actB   = (_Float16*)carve(ACT_ELEMS * 2);
  _Float16* wbuf   = (_Float16*)carve((size_t)512 * 4608 * 2);
  float* clsbuf    = (float*)carve((size_t)18 * PF * 4);
  float* bboxbuf   = (float*)carve((size_t)36 * PF * 4);
  float* fg        = (float*)carve((size_t)NANCH * 4);
  float* pbox      = (float*)carve((size_t)NANCH * 16);
  float* keys      = (float*)carve((size_t)NSORT * 4);
  int* sidx        = (int*)carve((size_t)NSORT * 4);
  unsigned char* keep = (unsigned char*)carve(8192);

  // image -> f16
  {
    int n = 3 * 600 * 800;
    f32_to_f16_kernel<<<(n + 255) / 256, 256, 0, stream>>>(image, actA, n);
  }

  const int cfgCi[13] = {3, 64, 64, 128, 128, 256, 256, 256, 512, 512, 512, 512, 512};
  const int cfgCo[13] = {64, 64, 128, 128, 256, 256, 256, 512, 512, 512, 512, 512, 512};

  int H = 600, W = 800;
  _Float16* cur = actA;
  _Float16* nxt = actB;

  for (int i = 0; i < 13; ++i) {
    int Ci = cfgCi[i], Co = cfgCo[i];
    int K  = Ci * 9;
    int Kp = (K + 31) & ~31;
    int Cop = (Co + 31) & ~31;
    cvt_weights_kernel<<<(Cop * Kp + 255) / 256, 256, 0, stream>>>(vw[i], wbuf, Co, K, Cop, Kp);
    int P = H * W;
    dim3 grid((P + PX_PER_BLOCK - 1) / PX_PER_BLOCK, Cop / 32);
    float* o32 = (i == 12) ? out : nullptr;   // last layer also writes feats (f32)
    conv_wmma_kernel<<<grid, 256, 0, stream>>>(cur, wbuf, vb[i], nxt, o32,
                                               Ci, Co, H, W, K, Kp, 1, 1);
    _Float16* t = cur; cur = nxt; nxt = t;
    if (i == 1 || i == 3 || i == 6 || i == 9) {
      int Ho = H / 2, Wo = W / 2;
      int tot = Co * Ho * Wo;
      pool2_kernel<<<(tot + 255) / 256, 256, 0, stream>>>(cur, nxt, Co, H, W, Ho, Wo);
      t = cur; cur = nxt; nxt = t;
      H = Ho; W = Wo;
    }
  }
  // now H=37, W=50; cur = feats (f16)

  // RPN 3x3 conv 512->512 + ReLU
  {
    int K = 512 * 9, Kp = K, Cop = 512;
    cvt_weights_kernel<<<(Cop * Kp + 255) / 256, 256, 0, stream>>>(conv1_w, wbuf, 512, K, Cop, Kp);
    dim3 grid((PF + PX_PER_BLOCK - 1) / PX_PER_BLOCK, Cop / 32);
    conv_wmma_kernel<<<grid, 256, 0, stream>>>(cur, wbuf, conv1_b, nxt, nullptr,
                                               512, 512, HF, WF, K, Kp, 1, 1);
    _Float16* t = cur; cur = nxt; nxt = t;
  }
  // cls 1x1 (18 ch, no relu) -> f32
  {
    int K = 512, Kp = 512, Cop = 32;
    cvt_weights_kernel<<<(Cop * Kp + 255) / 256, 256, 0, stream>>>(score_w, wbuf, 18, K, Cop, Kp);
    dim3 grid((PF + PX_PER_BLOCK - 1) / PX_PER_BLOCK, Cop / 32);
    conv_wmma_kernel<<<grid, 256, 0, stream>>>(cur, wbuf, score_b, nullptr, clsbuf,
                                               512, 18, HF, WF, K, Kp, 0, 0);
  }
  // bbox 1x1 (36 ch, no relu) -> f32
  {
    int K = 512, Kp = 512, Cop = 64;
    cvt_weights_kernel<<<(Cop * Kp + 255) / 256, 256, 0, stream>>>(bbox_w, wbuf, 36, K, Cop, Kp);
    dim3 grid((PF + PX_PER_BLOCK - 1) / PX_PER_BLOCK, Cop / 32);
    conv_wmma_kernel<<<grid, 256, 0, stream>>>(cur, wbuf, bbox_b, nullptr, bboxbuf,
                                               512, 36, HF, WF, K, Kp, 0, 0);
  }

  fg_score_kernel<<<(NANCH + 255) / 256, 256, 0, stream>>>(clsbuf, fg);
  decode_kernel<<<NSORT / 256, 256, 0, stream>>>(fg, bboxbuf, keys, sidx, pbox);

  for (int k = 2; k <= NSORT; k <<= 1)
    for (int j = k >> 1; j > 0; j >>= 1)
      bitonic_kernel<<<NSORT / 256, 256, 0, stream>>>(keys, sidx, j, k);

  nms_kernel<<<1, 1024, 0, stream>>>(keys, sidx, pbox, keep);
  finalize_kernel<<<1, 64, 0, stream>>>(keep, sidx, pbox, out);
}
